// QuantizedThriftyNet_32590211842409
// MI455X (gfx1250) — matrix operand order repaired and checked by reading, not compiled, not run on hardware
//
#include <hip/hip_runtime.h>

typedef __attribute__((ext_vector_type(16))) _Float16 v16h;
typedef __attribute__((ext_vector_type(8)))  _Float16 v8h;
typedef __attribute__((ext_vector_type(8)))  float    v8f;
typedef unsigned int u32x4 __attribute__((ext_vector_type(4)));
typedef int          i32x4 __attribute__((ext_vector_type(4)));
typedef int          i32x8 __attribute__((ext_vector_type(8)));

#define CIN   256
#define NF    256
#define KTOT  (9*CIN)      // 2304
#define KSTEP 64           // K-depth per LDS stage (64 | 256 -> single tap per stage)
#define NCLS  100
#define EPSV  1e-5f
#define AS1 __attribute__((address_space(1)))
#define AS3 __attribute__((address_space(3)))

#define HAVE_TDM        __has_builtin(__builtin_amdgcn_tensor_load_to_lds)
#define HAVE_ASYNC      __has_builtin(__builtin_amdgcn_global_load_async_to_lds_b128)
#define HAVE_WAITASYNC  __has_builtin(__builtin_amdgcn_s_wait_asynccnt)
#define HAVE_TDM6       __has_include(<hip/amd_detail/amd_gfx1250_TDM.h>)

#if HAVE_TDM
#pragma message("CDNA5: conv B-staging uses TDM tensor_load_to_lds (TENSORcnt)")
#elif HAVE_ASYNC
#pragma message("CDNA5: conv B-staging uses global_load_async_to_lds_b128 (ASYNCcnt)")
#else
#pragma message("CDNA5: conv B-staging uses generic load/ds_store fallback")
#endif

// ---------------- weight quantization ----------------

__global__ void k_zero_u32(unsigned* p) {
    if (threadIdx.x == 0 && blockIdx.x == 0) *p = 0u;
}

__global__ void k_zero_f32(float* p, int n) {
    int i = blockIdx.x * 256 + threadIdx.x;
    if (i < n) p[i] = 0.f;
}

__global__ void k_absmax(const float* __restrict__ w, int n, unsigned* __restrict__ out) {
    __shared__ unsigned sm[256];
    int i = blockIdx.x * 256 + threadIdx.x;
    float v = (i < n) ? fabsf(w[i]) : 0.f;
    sm[threadIdx.x] = __float_as_uint(v);   // abs floats: uint order == float order
    __syncthreads();
    for (int s = 128; s > 0; s >>= 1) {
        if (threadIdx.x < s) {
            unsigned a = sm[threadIdx.x], b = sm[threadIdx.x + s];
            sm[threadIdx.x] = (a > b) ? a : b;
        }
        __syncthreads();
    }
    if (threadIdx.x == 0) atomicMax(out, sm[0]);
}

// repack OIHW f32 -> f16 B[cout][ (ky*3+kx)*256 + cin ], quantized (8-bit STE forward)
__global__ void k_quant(const float* __restrict__ w, const unsigned* __restrict__ absmax,
                        _Float16* __restrict__ wq, int n) {
    int i = blockIdx.x * 256 + threadIdx.x;
    if (i >= n) return;
    float s = __uint_as_float(*absmax) * (1.0f / 127.0f);
    int kx = i % 3;
    int ky = (i / 3) % 3;
    int c  = (i / 9) % CIN;
    int o  = i / (9 * CIN);
    float q = rintf(w[i] / s) * s;   // RNE == jnp.round
    wq[(size_t)o * KTOT + (ky * 3 + kx) * CIN + c] = (_Float16)q;
}

// x (16,3,64,64) NCHW f32 -> (16,64,64,256) NHWC f16, channels 3..255 zero
__global__ void k_pad(const float* __restrict__ x, _Float16* __restrict__ dst) {
    int i = blockIdx.x * 256 + threadIdx.x;   // 16*64*64*256 total
    int c   = i & 255;
    int pos = i >> 8;                          // n*4096 + y*64 + x
    int xx = pos & 63, yy = (pos >> 6) & 63, nn = pos >> 12;
    float v = 0.f;
    if (c < 3) v = x[((nn * 3 + c) * 64 + yy) * 64 + xx];
    dst[i] = (_Float16)v;
}

// ---------------- TDM / async staging helpers ----------------

#if HAVE_TDM
// One 2D TDM descriptor: tile = KSTEP (K halves) x 128 (cout rows), 2B elems,
// row stride KTOT elems.  D# layout per CDNA5 ISA 8.3/8.4.
static __device__ __forceinline__ void tdm_load_b_tile(const _Float16* gbase, unsigned ldsOff) {
    unsigned long long ga = (unsigned long long)gbase;
    u32x4 g0;
    g0[0] = 1u;                                              // count=1, user mode
    g0[1] = ldsOff;                                          // lds_addr (bytes)
    g0[2] = (unsigned)(ga & 0xFFFFFFFFull);                  // global_addr lo
    g0[3] = (unsigned)((ga >> 32) & 0x01FFFFFFull) | (2u << 30); // addr hi | type=2
    i32x8 g1;
    g1[0] = 1 << 16;          // workgroup_mask=0, data_size=1 (2 bytes)
    g1[1] = KSTEP << 16;      // tensor_dim0 = KSTEP (lo16)
    g1[2] = 128 << 16;        // tensor_dim0 hi=0 | tensor_dim1 = 128 (lo16)
    g1[3] = KSTEP << 16;      // tensor_dim1 hi=0 | tile_dim0 = KSTEP
    g1[4] = 128;              // tile_dim1 = 128, tile_dim2 = 0
    g1[5] = KTOT;             // tensor_dim0_stride lo32
    g1[6] = 0;
    g1[7] = 0;
    i32x4 z4 = {0, 0, 0, 0};
#if HAVE_TDM6
    i32x8 z8 = {0, 0, 0, 0, 0, 0, 0, 0};
    __builtin_amdgcn_tensor_load_to_lds(g0, g1, z4, z4, z8, 0);
#else
    __builtin_amdgcn_tensor_load_to_lds(g0, g1, z4, z4, 0);
#endif
}
#endif

// ---------------- implicit-GEMM conv + relu + alpha combine ----------------
// pre[m][cout] = a0*relu(conv) + a2*h1 + a3*h2 + a4*h3   (NHWC, m = n*S*S + y*S + x)
// Block: 8 waves, tile 128(M) x 128(N); wave = 32x64 via 2x4 wmma accumulators.
// B tile (128 couts x KSTEP K-halves) staged in LDS via TDM, double buffered.
// 16 v_wmma per stage, 36 stages.

__global__ __launch_bounds__(256)
void k_conv_wmma(const _Float16* __restrict__ act,   // conv input = hist[3], NHWC f16
                 const _Float16* __restrict__ wq,    // B[256][2304]
                 const _Float16* __restrict__ h1,
                 const _Float16* __restrict__ h2,
                 const _Float16* __restrict__ h3,
                 const float* __restrict__ al,       // alpha row (5 floats)
                 _Float16* __restrict__ pre,
                 int S, int live1, int live2)
{
    __shared__ __align__(64) _Float16 Bs[2][128 * KSTEP];   // 2 x 16KB, double buffered

    const int lane = threadIdx.x & 31;
    const int wave = threadIdx.x >> 5;
    const int wm = wave & 3;        // 4 waves along M -> block M tile 128
    const int wn = wave >> 2;       // 2 waves along N -> block N tile 128
    const int lm   = lane & 15;
    const int hi16 = lane >> 4;     // 0/1: which K-half this lane holds
    const int klo  = hi16 * 16;     // A: local K offset (lanes 16-31 hold K 16..31)
    const int kbo  = hi16 * 16;     // B: local K offset (lanes 16-31 hold K 16..31)
    const int nb0  = blockIdx.y * 128;           // block's first cout
    const int SS = S * S;

    int ni[2], yi[2], xi[2];
    #pragma unroll
    for (int i = 0; i < 2; i++) {
        int m = blockIdx.x * 128 + wm * 32 + i * 16 + lm;
        ni[i] = m / SS;
        int r = m - ni[i] * SS;
        yi[i] = r / S;
        xi[i] = r - yi[i] * S;
    }

    v8f acc[2][4];
    const v8f zf = {0.f,0.f,0.f,0.f,0.f,0.f,0.f,0.f};
    #pragma unroll
    for (int i = 0; i < 2; i++)
        #pragma unroll
        for (int j = 0; j < 4; j++) acc[i][j] = zf;

#if HAVE_TDM
    unsigned bOff[2];
    bOff[0] = (unsigned)(unsigned long long)(AS3 _Float16*)&Bs[0][0];
    bOff[1] = (unsigned)(unsigned long long)(AS3 _Float16*)&Bs[1][0];
    if (wave == 0) tdm_load_b_tile(wq + (size_t)nb0 * KTOT + 0, bOff[0]);
#else
    // cooperative stage: 128 rows x 128B = 1024 x 16B chunks, 4 per thread
    for (int l = threadIdx.x; l < 1024; l += 256) {
        int row = l >> 3, q = l & 7;
        const _Float16* g = wq + (size_t)(nb0 + row) * KTOT + 0 + q * 8;
        _Float16* d = &Bs[0][row * KSTEP + q * 8];
#if HAVE_ASYNC
        __builtin_amdgcn_global_load_async_to_lds_b128(
            (AS1 const void*)g, (AS3 void*)d, 0, 0);
#else
        *(v8h*)d = *(const v8h*)g;
#endif
    }
#endif

    int cur = 0;
    for (int kk = 0; kk < KTOT; kk += KSTEP) {
        const int more = (kk + KSTEP) < KTOT;

        // issue next stage into the other buffer
        if (more) {
#if HAVE_TDM
            if (wave == 0)
                tdm_load_b_tile(wq + (size_t)nb0 * KTOT + (kk + KSTEP), bOff[cur ^ 1]);
#else
            for (int l = threadIdx.x; l < 1024; l += 256) {
                int row = l >> 3, q = l & 7;
                const _Float16* g = wq + (size_t)(nb0 + row) * KTOT + (kk + KSTEP) + q * 8;
                _Float16* d = &Bs[cur ^ 1][row * KSTEP + q * 8];
#if HAVE_ASYNC
                __builtin_amdgcn_global_load_async_to_lds_b128(
                    (AS1 const void*)g, (AS3 void*)d, 0, 0);
#else
                *(v8h*)d = *(const v8h*)g;
#endif
            }
#endif
        }

        // wait for current stage, publish to all waves
#if HAVE_TDM
        if (wave == 0) {
            if (more) __builtin_amdgcn_s_wait_tensorcnt(1);
            else      __builtin_amdgcn_s_wait_tensorcnt(0);
        }
#elif HAVE_ASYNC
#if HAVE_WAITASYNC
        if (more) __builtin_amdgcn_s_wait_asynccnt(4);
        else      __builtin_amdgcn_s_wait_asynccnt(0);
#else
        if (more) asm volatile("s_wait_asynccnt 0x4" ::: "memory");
        else      asm volatile("s_wait_asynccnt 0x0" ::: "memory");
#endif
#endif
        __syncthreads();

        int tap  = kk >> 8;                 // 0..8 -> (ky,kx); KSTEP | 256
        int dy   = tap / 3 - 1;
        int dx   = tap - (tap / 3) * 3 - 1;
        int cin0 = kk & 255;

        // A fragments for both 32-K sub-chunks: one bounds check, 4x16B loads
        v16h a[2][2];                       // [m-subtile][k-subchunk]
        #pragma unroll
        for (int i = 0; i < 2; i++) {
            v8h c0 = {0,0,0,0,0,0,0,0}, c1 = c0, c2 = c0, c3 = c0;
            int yy = yi[i] + dy, xx = xi[i] + dx;
            if (yy >= 0 && yy < S && xx >= 0 && xx < S) {
                const _Float16* p = act + (((size_t)ni[i] * S + yy) * S + xx) * CIN + cin0 + klo;
                c0 = *(const v8h*)p;          // K = klo..klo+7
                c1 = *(const v8h*)(p + 8);    // K = 8+klo..
                c2 = *(const v8h*)(p + 32);   // K = 32+klo..
                c3 = *(const v8h*)(p + 40);   // K = 40+klo..
                if (i == 0) __builtin_prefetch(p + KSTEP, 0, 0);
            }
            a[i][0] = __builtin_shufflevector(c0, c1, 0,1,2,3,4,5,6,7,8,9,10,11,12,13,14,15);
            a[i][1] = __builtin_shufflevector(c2, c3, 0,1,2,3,4,5,6,7,8,9,10,11,12,13,14,15);
        }

        // B fragments from LDS: lane = column, 16 contiguous K halves
        v16h b[2][4];                       // [k-subchunk][n-subtile]
        #pragma unroll
        for (int u = 0; u < 2; u++)
            #pragma unroll
            for (int j = 0; j < 4; j++) {
                int row = wn * 64 + j * 16 + lm;              // block-relative cout
                b[u][j] = *(const v16h*)(&Bs[cur][row * KSTEP + u * 32 + kbo]);
            }

        #pragma unroll
        for (int u = 0; u < 2; u++)
            #pragma unroll
            for (int i = 0; i < 2; i++)
                #pragma unroll
                for (int j = 0; j < 4; j++)
                    acc[i][j] = __builtin_amdgcn_wmma_f32_16x16x32_f16(
                        false, a[i][u], false, b[u][j], (short)0, acc[i][j], false, false);

        __syncthreads();     // all waves done reading Bs[cur] before it is re-staged
        cur ^= 1;
    }

    // epilogue: relu + alpha combine, f16 store (C layout: c[r] = C[r + 8*hi16][lm])
    const float a0 = al[0], a2 = al[2], a3 = al[3], a4 = al[4];
    #pragma unroll
    for (int i = 0; i < 2; i++) {
        int mb = blockIdx.x * 128 + wm * 32 + i * 16 + hi16 * 8;
        #pragma unroll
        for (int j = 0; j < 4; j++) {
            int nc = nb0 + wn * 64 + j * 16 + lm;
            #pragma unroll
            for (int r = 0; r < 8; r++) {
                size_t idx = (size_t)(mb + r) * NF + nc;
                float v = a0 * fmaxf(acc[i][j][r], 0.f);
                if (live1) v += a2 * (float)h1[idx];
                if (live2) v += a3 * (float)h2[idx];
                v += a4 * (float)h3[idx];
                pre[idx] = (_Float16)v;
            }
        }
    }
}

// ---------------- batchnorm ----------------

__global__ void k_bn_stats(const _Float16* __restrict__ pre,
                           float* __restrict__ sum, float* __restrict__ sumsq, int M) {
    int c = threadIdx.x;                       // 256 channels
    int rpb = M / gridDim.x;
    int r0 = blockIdx.x * rpb;
    float s = 0.f, q = 0.f;
    for (int r = 0; r < rpb; r++) {
        float v = (float)pre[(size_t)(r0 + r) * NF + c];
        s += v; q += v * v;
    }
    atomicAdd(&sum[c], s);
    atomicAdd(&sumsq[c], q);
}

__global__ void k_bn_fin(const float* __restrict__ sum, const float* __restrict__ sumsq,
                         const float* __restrict__ gamma, const float* __restrict__ beta,
                         float* __restrict__ scale, float* __restrict__ shift, int M) {
    int c = threadIdx.x;
    float inv  = 1.0f / (float)M;
    float mean = sum[c] * inv;
    float var  = sumsq[c] * inv - mean * mean;   // biased variance
    float sc   = gamma[c] * rsqrtf(var + EPSV);
    scale[c] = sc;
    shift[c] = beta[c] - mean * sc;
}

__global__ void k_bn_apply(const _Float16* __restrict__ pre,
                           const float* __restrict__ scale, const float* __restrict__ shift,
                           _Float16* __restrict__ dst, int total8) {
    int i = blockIdx.x * 256 + threadIdx.x;
    if (i >= total8) return;
    size_t base = (size_t)i * 8;
    int c0 = (int)(base & 255);
    v8h p = *(const v8h*)(pre + base);
    v8h o;
    #pragma unroll
    for (int e = 0; e < 8; e++)
        o[e] = (_Float16)((float)p[e] * scale[c0 + e] + shift[c0 + e]);
    *(v8h*)(dst + base) = o;
}

// ---------------- maxpool 2x2, NHWC f16, 8 channels/thread ----------------

__global__ void k_pool(const _Float16* __restrict__ src, _Float16* __restrict__ dst,
                       int So, int total8) {
    int i = blockIdx.x * 256 + threadIdx.x;
    if (i >= total8) return;
    size_t idx = (size_t)i * 8;
    int c   = (int)(idx & 255);
    int pos = (int)(idx >> 8);
    int x = pos % So;
    int y = (pos / So) % So;
    int n = pos / (So * So);
    int Si = So * 2;
    size_t b = (((size_t)n * Si + 2 * y) * Si + 2 * x) * NF + c;
    v8h v0 = *(const v8h*)(src + b);
    v8h v1 = *(const v8h*)(src + b + NF);
    v8h v2 = *(const v8h*)(src + b + (size_t)Si * NF);
    v8h v3 = *(const v8h*)(src + b + (size_t)Si * NF + NF);
    v8h o;
    #pragma unroll
    for (int e = 0; e < 8; e++) {
        float m = fmaxf(fmaxf((float)v0[e], (float)v1[e]),
                        fmaxf((float)v2[e], (float)v3[e]));
        o[e] = (_Float16)m;
    }
    *(v8h*)(dst + idx) = o;
}

// ---------------- global max + linear head ----------------

__global__ void k_head(const _Float16* __restrict__ act, const float* __restrict__ ow,
                       const float* __restrict__ ob, float* __restrict__ out, int S) {
    __shared__ float feat[NF];
    int n = blockIdx.x, c = threadIdx.x;
    int P = S * S;
    float m = -1e30f;
    for (int p = 0; p < P; p++)
        m = fmaxf(m, (float)act[((size_t)n * P + p) * NF + c]);
    feat[c] = m;
    __syncthreads();
    for (int k = c; k < NCLS; k += 256) {
        float s = ob[k];
        for (int cc = 0; cc < NF; cc++) s += feat[cc] * ow[k * NF + cc];
        out[n * NCLS + k] = s;
    }
}

// ---------------- host orchestration (graph-capturable) ----------------

extern "C" void kernel_launch(void* const* d_in, const int* in_sizes, int n_in,
                              void* d_out, int out_size, void* d_ws, size_t ws_size,
                              hipStream_t stream) {
    const float* x      = (const float*)d_in[0];
    const float* conv_w = (const float*)d_in[1];
    const float* alpha  = (const float*)d_in[2];
    const float* gamma  = (const float*)d_in[3];
    const float* beta   = (const float*)d_in[4];
    const float* ow     = (const float*)d_in[5];
    const float* ob     = (const float*)d_in[6];
    float* out = (float*)d_out;

    char* ws = (char*)d_ws;
    size_t off = 0;
    auto carve = [&](size_t bytes) -> void* {
        off = (off + 255) & ~(size_t)255;
        void* p = ws + off;
        off += bytes;
        return p;
    };

    unsigned* absmax = (unsigned*)carve(4);
    float* sum   = (float*)carve(NF * 4);
    float* sumsq = (float*)carve(NF * 4);
    float* scale = (float*)carve(NF * 4);
    float* shift = (float*)carve(NF * 4);
    _Float16* wq = (_Float16*)carve((size_t)KTOT * NF * 2);
    const size_t ACT = (size_t)16 * 64 * 64 * 256;     // max activation elems
    _Float16* pre = (_Float16*)carve(ACT * 2);
    _Float16* slot[6];
    for (int i = 0; i < 6; i++) slot[i] = (_Float16*)carve(ACT * 2);

    bool used[6] = {false,false,false,false,false,false};
    auto salloc = [&]() -> int {
        for (int i = 0; i < 6; i++) if (!used[i]) { used[i] = true; return i; }
        return 0;
    };

    // quantize + repack weights
    const int nW = NF * CIN * 9;
    k_zero_u32<<<1, 1, 0, stream>>>(absmax);
    k_absmax<<<(nW + 255) / 256, 256, 0, stream>>>(conv_w, nW, absmax);
    k_quant<<<(nW + 255) / 256, 256, 0, stream>>>(conv_w, absmax, wq, nW);

    // hist[1..3]; hist[0] is always None in the reference
    int h1 = -1, h2 = -1, h3 = salloc();
    k_pad<<<(int)(ACT / 256), 256, 0, stream>>>(x, slot[h3]);

    int S = 64;
    for (int t = 0; t < 20; t++) {
        int M = 16 * S * S;                 // always divisible by 128

        k_zero_f32<<<1, 256, 0, stream>>>(sum, NF);
        k_zero_f32<<<1, 256, 0, stream>>>(sumsq, NF);

        dim3 cg(M / 128, 2);
        k_conv_wmma<<<cg, 256, 0, stream>>>(
            slot[h3], wq,
            (h1 >= 0) ? slot[h1] : slot[h3],
            (h2 >= 0) ? slot[h2] : slot[h3],
            slot[h3],
            alpha + t * 5, pre, S, (h1 >= 0) ? 1 : 0, (h2 >= 0) ? 1 : 0);

        k_bn_stats<<<256, 256, 0, stream>>>(pre, sum, sumsq, M);
        k_bn_fin<<<1, 256, 0, stream>>>(sum, sumsq, gamma + t * NF, beta + t * NF,
                                        scale, shift, M);

        int na = salloc();
        int total8 = (int)((size_t)M * NF / 8);
        k_bn_apply<<<(total8 + 255) / 256, 256, 0, stream>>>(
            pre, scale, shift, slot[na], total8);

        if (h1 >= 0) used[h1] = false;      // hist shift: drop hist[1]
        h1 = h2; h2 = h3; h3 = na;

        if (t % 5 == 4) {                   // pool all live history entries
            int So = S / 2;
            int tot8 = 16 * So * So * NF / 8;
            int* hs[3] = {&h1, &h2, &h3};
            for (int q = 0; q < 3; q++) {
                int cur = *hs[q];
                if (cur < 0) continue;
                int d = salloc();
                k_pool<<<(tot8 + 255) / 256, 256, 0, stream>>>(slot[cur], slot[d], So, tot8);
                used[cur] = false;
                *hs[q] = d;
            }
            S = So;
        }
    }

    k_head<<<16, 256, 0, stream>>>(slot[h3], ow, ob, out, S);
}
